// FunctionalConnectivityEncoder_40931038331171
// MI455X (gfx1250) — matrix operand order repaired and checked by reading, not compiled, not run on hardware
//
#include <hip/hip_runtime.h>

// ---------------------------------------------------------------------------
// MI455X (gfx1250) CDNA5 implementation. Wave32. All heavy GEMMs use
// v_wmma_f32_16x16x32_f16 with f32 accumulation.
// Preconditions exploited: every GEMM here has M % 128 == 0, Nc % 64 == 0,
// K % 32 == 0 -> all tiles are full, no bounds guards anywhere.
// ---------------------------------------------------------------------------

typedef __attribute__((ext_vector_type(16))) _Float16 v16h;
typedef __attribute__((ext_vector_type(8)))  _Float16 v8h;
typedef __attribute__((ext_vector_type(8)))  float    v8f;

#define BDIM 256
#define LEAKY 0.2f

__device__ __forceinline__ v16h ld16(const _Float16* p0, const _Float16* p1) {
    v8h lo = *(const v8h*)p0;
    v8h hi = *(const v8h*)p1;
    v16h r;
#pragma unroll
    for (int i = 0; i < 8; ++i) { r[i] = lo[i]; r[i + 8] = hi[i]; }
    return r;
}

__device__ __forceinline__ float wave_max(float v) {
#pragma unroll
    for (int o = 16; o > 0; o >>= 1) v = fmaxf(v, __shfl_xor(v, o, 32));
    return v;
}
__device__ __forceinline__ float wave_sum(float v) {
#pragma unroll
    for (int o = 16; o > 0; o >>= 1) v += __shfl_xor(v, o, 32);
    return v;
}

#define WMMA_F16(A, B, C) \
    __builtin_amdgcn_wmma_f32_16x16x32_f16(false, (A), false, (B), (short)0, (C), false, false)

// ---------------------------------------------------------------------------
// Staging helpers for the double-buffered GEMM (regs -> LDS, f32 -> f16)
// ---------------------------------------------------------------------------
__device__ __forceinline__ void gemm_load_tile(
    const float* __restrict__ A, const float* __restrict__ Bm,
    int lda, int ldb, int i0, int j0, int k0, int tid,
    float4* fa, float4* fb)
{
#pragma unroll
    for (int rep = 0; rep < 4; ++rep) {           // A: 128x32 floats
        int q = rep * BDIM + tid;                 // 0..1023 float4s
        int row = q >> 3, c4 = q & 7;
        fa[rep] = *(const float4*)(A + (long)(i0 + row) * lda + k0 + c4 * 4);
    }
#pragma unroll
    for (int rep = 0; rep < 2; ++rep) {           // B: 32x64 floats
        int q = rep * BDIM + tid;                 // 0..511 float4s
        int kk = q >> 4, n4 = q & 15;
        fb[rep] = *(const float4*)(Bm + (long)(k0 + kk) * ldb + j0 + n4 * 4);
    }
}

__device__ __forceinline__ void gemm_store_tile(
    _Float16 (*__restrict__ As)[40], _Float16 (*__restrict__ Bs)[40],
    int tid, const float4* fa, const float4* fb)
{
#pragma unroll
    for (int rep = 0; rep < 4; ++rep) {
        int q = rep * BDIM + tid;
        int row = q >> 3, c4 = q & 7;
        As[row][c4 * 4 + 0] = (_Float16)fa[rep].x;
        As[row][c4 * 4 + 1] = (_Float16)fa[rep].y;
        As[row][c4 * 4 + 2] = (_Float16)fa[rep].z;
        As[row][c4 * 4 + 3] = (_Float16)fa[rep].w;
    }
#pragma unroll
    for (int rep = 0; rep < 2; ++rep) {
        int q = rep * BDIM + tid;
        int kk = q >> 4, n4 = q & 15;
        Bs[n4 * 4 + 0][kk] = (_Float16)fb[rep].x;
        Bs[n4 * 4 + 1][kk] = (_Float16)fb[rep].y;
        Bs[n4 * 4 + 2][kk] = (_Float16)fb[rep].z;
        Bs[n4 * 4 + 3][kk] = (_Float16)fb[rep].w;
    }
}

// ---------------------------------------------------------------------------
// Tiled GEMM:  C = act( rowscale * (A @ B) + bias )
// A: MxK (lda), B: KxNc (ldb), C: MxNc (ldc). f32 in/out, f16 WMMA compute.
// Block tile 128x64, 8 waves (4x2), each wave a 32x32 sub-tile:
// 2 A-frags x 2 B-frags -> 4 v_wmma per K-step. Double-buffered LDS with
// register pipelining: one barrier per K-step, staging loads issue as a
// clause overlapped with WMMA compute. Batched via blockIdx.z.
// ---------------------------------------------------------------------------
__global__ __launch_bounds__(BDIM) void wmma_gemm(
    const float* __restrict__ A, const float* __restrict__ Bm,
    const float* __restrict__ bias, float* __restrict__ C,
    int K, int lda, int ldb, int ldc,
    long sA, long sB, long sC,
    const float* __restrict__ rowscale, long sRS, int act)
{
    __shared__ _Float16 As[2][128][40];  // [m][k], row stride 80B (16B aligned)
    __shared__ _Float16 Bs[2][64][40];   // [n][k] (transposed on stage)

    const int z = blockIdx.z;
    A += (long)z * sA;  Bm += (long)z * sB;  C += (long)z * sC;
    if (rowscale) rowscale += (long)z * sRS;

    const int tid  = threadIdx.x;
    const int wave = tid >> 5, lane = tid & 31;
    const int wr = wave >> 1, wc = wave & 1;       // 4x2 wave grid
    const int laneN = lane & 15, hl = lane >> 4;
    const int i0 = blockIdx.x * 128, j0 = blockIdx.y * 64;

    v8f c00 = {}, c01 = {}, c10 = {}, c11 = {};
    float4 fa[4], fb[2];

    // prologue: stage tile 0 into buffer 0
    gemm_load_tile(A, Bm, lda, ldb, i0, j0, 0, tid, fa, fb);
    gemm_store_tile(As[0], Bs[0], tid, fa, fb);
    __syncthreads();

    int p = 0;
    for (int k0 = 0; k0 < K; k0 += 32) {
        const bool more = (k0 + 32 < K);
        if (more) {
            gemm_load_tile(A, Bm, lda, ldb, i0, j0, k0 + 32, tid, fa, fb);
            if (k0 + 64 < K)   // prefetch 2 tiles ahead (global_prefetch_b8)
                __builtin_prefetch(A + (long)(i0 + (tid >> 1)) * lda + k0 + 64 + (tid & 1) * 16, 0, 1);
        }

        // A frags: rows wr*32 + {laneN, 16+laneN}; halves = K {8*hl.., +16}
        const _Float16* ar0 = &As[p][wr * 32 + laneN][0];
        const _Float16* ar1 = &As[p][wr * 32 + 16 + laneN][0];
        v16h a0 = ld16(ar0 + 8 * hl, ar0 + 8 * hl + 16);
        v16h a1 = ld16(ar1 + 8 * hl, ar1 + 8 * hl + 16);
        // B frags: cols wc*32 + {laneN, 16+laneN}; halves = K {8*hl.., +16}
        const _Float16* b0p = &Bs[p][wc * 32 + laneN][8 * hl];
        const _Float16* b1p = &Bs[p][wc * 32 + 16 + laneN][8 * hl];
        v16h b0 = ld16(b0p, b0p + 16);
        v16h b1 = ld16(b1p, b1p + 16);

        c00 = WMMA_F16(a0, b0, c00);
        c01 = WMMA_F16(a0, b1, c01);
        c10 = WMMA_F16(a1, b0, c10);
        c11 = WMMA_F16(a1, b1, c11);

        if (more) {
            gemm_store_tile(As[p ^ 1], Bs[p ^ 1], tid, fa, fb);
            __syncthreads();
            p ^= 1;
        }
    }

    // epilogue: C-frag row m = r + 8*hl, col = laneN
    const int colA = j0 + wc * 32 + laneN;
    const float biasA = bias ? bias[colA] : 0.f;
    const float biasB = bias ? bias[colA + 16] : 0.f;
#pragma unroll
    for (int r = 0; r < 8; ++r) {
        int row0 = i0 + wr * 32 + r + 8 * hl;
        int row1 = row0 + 16;
        float rs0 = rowscale ? rowscale[row0] : 1.f;
        float rs1 = rowscale ? rowscale[row1] : 1.f;
        float v00 = c00[r] * rs0 + biasA;
        float v01 = c01[r] * rs0 + biasB;
        float v10 = c10[r] * rs1 + biasA;
        float v11 = c11[r] * rs1 + biasB;
        if (act == 1) {
            v00 = fmaxf(v00, 0.f); v01 = fmaxf(v01, 0.f);
            v10 = fmaxf(v10, 0.f); v11 = fmaxf(v11, 0.f);
        } else if (act == 2) {
            v00 = tanhf(v00); v01 = tanhf(v01);
            v10 = tanhf(v10); v11 = tanhf(v11);
        }
        C[(long)row0 * ldc + colA]      = v00;
        C[(long)row0 * ldc + colA + 16] = v01;
        C[(long)row1 * ldc + colA]      = v10;
        C[(long)row1 * ldc + colA + 16] = v11;
    }
}

// ---------------------------------------------------------------------------
// Pack gat_W[l] [H,128,32] -> Wcat [128, H*32] so h = x @ Wcat
// ---------------------------------------------------------------------------
__global__ void pack_wcat(const float* __restrict__ gatW, float* __restrict__ Wcat) {
    int t = blockIdx.x * BDIM + threadIdx.x;   // 16384
    int i = t >> 7, c = t & 127;
    int h = c >> 5, dd = c & 31;
    Wcat[i * 128 + c] = gatW[((long)h * 128 + i) * 32 + dd];
}

// ---------------------------------------------------------------------------
// e_src/e_dst = h . a_src / a_dst  per (b,h,n)
// ---------------------------------------------------------------------------
__global__ void gat_scores(const float* __restrict__ hbuf, const float* __restrict__ ga,
                           float* __restrict__ esrc, float* __restrict__ edst) {
    int t = blockIdx.x * BDIM + threadIdx.x;   // B*H*N = 32768
    int n = t & 1023, bh = t >> 10;
    int h = bh & 3, b = bh >> 2;
    const float* hr = hbuf + ((long)b * 1024 + n) * 128 + h * 32;
    const float* av = ga + h * 64;
    float s = 0.f, d = 0.f;
#pragma unroll
    for (int k = 0; k < 32; ++k) { float hv = hr[k]; s += hv * av[k]; d += hv * av[32 + k]; }
    esrc[t] = s; edst[t] = d;
}

// ---------------------------------------------------------------------------
// Per-row softmax stats: m_i = max_j f(s_i+d_j), invZ = 1/sum exp. Wave/row.
// ---------------------------------------------------------------------------
__global__ __launch_bounds__(BDIM) void gat_stats(
    const float* __restrict__ fc, const float* __restrict__ esrc,
    const float* __restrict__ edst, float* __restrict__ mrow, float* __restrict__ invZ)
{
    int wave = threadIdx.x >> 5, lane = threadIdx.x & 31;
    int t = blockIdx.x * 8 + wave;             // 0..32767
    int n = t & 1023, bh = t >> 10, b = bh >> 2;
    const float* dv = edst + ((long)bh << 10);
    const float* fr = fc + ((long)b * 1024 + n) * 1024;
    float s = esrc[t];
    float mx = -__builtin_inff();
    for (int j = lane; j < 1024; j += 32) {
        float v = s + dv[j];
        v = v > 0.f ? v : LEAKY * v;
        if (fr[j] == 0.f) v = -__builtin_inff();
        mx = fmaxf(mx, v);
    }
    mx = wave_max(mx);
    float sum = 0.f;
    for (int j = lane; j < 1024; j += 32) {
        float v = s + dv[j];
        v = v > 0.f ? v : LEAKY * v;
        if (fr[j] != 0.f) sum += __expf(v - mx);
    }
    sum = wave_sum(sum);
    if (lane == 0) { mrow[t] = mx; invZ[t] = 1.f / sum; }
}

// ---------------------------------------------------------------------------
// Fused GAT attention: hp = softmax(LeakyReLU(s_i+d_j), mask) @ h, then the
// torch-faithful scrambled reshape [B,H,N,d]->[B,N,H*d], relu + residual to x.
// Grid: (N/128, H, B). 8 waves; each wave a 16-row i-tile, streams all j.
// Exp-score A-fragment built directly in the WMMA register layout.
// ---------------------------------------------------------------------------
#define JCH 256
__global__ __launch_bounds__(BDIM) void gat_attend(
    const float* __restrict__ fc, const float* __restrict__ hbuf,
    const float* __restrict__ esrc, const float* __restrict__ edst,
    const float* __restrict__ mrow, const float* __restrict__ invZ,
    float* __restrict__ x)
{
    __shared__ _Float16 hs[32][JCH + 8];   // hs[dd][j], row stride 528B
    __shared__ float dsh[JCH];

    const int b = blockIdx.z, h = blockIdx.y;
    const int tid = threadIdx.x, wave = tid >> 5, lane = tid & 31;
    const int m = lane & 15, hl = lane >> 4;
    const int ibase = blockIdx.x * 128 + wave * 16;
    const int i = ibase + m;
    const long bhN = ((long)b * 4 + h) * 1024;
    const float si = esrc[bhN + i];
    const float mi = mrow[bhN + i];
    const float* fcrow = fc + ((long)b * 1024 + i) * 1024;

    v8f c0 = {}, c1 = {};

    for (int jc = 0; jc < 1024; jc += JCH) {
        // stage h[b,h, jc..jc+JCH, :32] transposed into LDS as f16 (float4 loads)
        float4 fh[8];
#pragma unroll
        for (int rep = 0; rep < 8; ++rep) {
            int q = rep * BDIM + tid;          // 0..2047 float4s
            int j = q >> 3, c4 = q & 7;
            fh[rep] = *(const float4*)(hbuf + ((long)b * 1024 + jc + j) * 128 + h * 32 + c4 * 4);
        }
        float dv = edst[bhN + jc + tid];       // JCH == BDIM
#pragma unroll
        for (int rep = 0; rep < 8; ++rep) {
            int q = rep * BDIM + tid;
            int j = q >> 3, c4 = q & 7;
            hs[c4 * 4 + 0][j] = (_Float16)fh[rep].x;
            hs[c4 * 4 + 1][j] = (_Float16)fh[rep].y;
            hs[c4 * 4 + 2][j] = (_Float16)fh[rep].z;
            hs[c4 * 4 + 3][j] = (_Float16)fh[rep].w;
        }
        dsh[tid] = dv;
        __syncthreads();

#pragma unroll
        for (int jt = 0; jt < JCH; jt += 32) {
            v16h a;
#pragma unroll
            for (int hf = 0; hf < 16; ++hf) {
                int Kk = (hf & 7) + 8 * hl + 16 * (hf >> 3);
                int jl = jt + Kk;
                float v = si + dsh[jl];
                v = v > 0.f ? v : LEAKY * v;
                float fcv = fcrow[jc + jl];
                float p = (fcv == 0.f) ? 0.f : __expf(v - mi);
                a[hf] = (_Float16)p;
            }
            const _Float16* b0p = &hs[m][jt + 8 * hl];
            const _Float16* b1p = &hs[16 + m][jt + 8 * hl];
            v16h b0 = ld16(b0p, b0p + 16);
            v16h b1 = ld16(b1p, b1p + 16);
            c0 = WMMA_F16(a, b0, c0);
            c1 = WMMA_F16(a, b1, c1);
        }
        __syncthreads();
    }

    // scale by 1/Z; scrambled reshape hp[b,h,n,dd] -> x[b, h*256+n/4, (n%4)*32+dd]
#pragma unroll
    for (int r = 0; r < 8; ++r) {
        int ii = ibase + r + 8 * hl;
        float iz = invZ[bhN + ii];
        float v0 = c0[r] * iz, v1 = c1[r] * iz;
        int np = h * 256 + (ii >> 2);
        int cb = (ii & 3) * 32 + m;
        long o0 = ((long)b * 1024 + np) * 128 + cb;
        long o1 = o0 + 16;
        x[o0] = fmaxf(v0, 0.f) + x[o0];
        x[o1] = fmaxf(v1, 0.f) + x[o1];
    }
}

// ---------------------------------------------------------------------------
// invrs[b,n] = 1 / max(sum_j fc[b,n,j], 1e-6) ; one wave per row
// ---------------------------------------------------------------------------
__global__ __launch_bounds__(BDIM) void row_inv(const float* __restrict__ fc, float* __restrict__ inv) {
    int wave = threadIdx.x >> 5, lane = threadIdx.x & 31;
    long row = (long)blockIdx.x * 8 + wave;    // 8192 rows
    const float* fr = fc + row * 1024;
    float s = 0.f;
    for (int j = lane; j < 1024; j += 32) s += fr[j];
    s = wave_sum(s);
    if (lane == 0) inv[row] = 1.f / fmaxf(s, 1e-6f);
}

// ---------------------------------------------------------------------------
// LayerNorm(128) + relu, in place; one wave per row
// ---------------------------------------------------------------------------
__global__ __launch_bounds__(BDIM) void ln_relu(float* __restrict__ ms,
                                                const float* __restrict__ g,
                                                const float* __restrict__ bb) {
    int wave = threadIdx.x >> 5, lane = threadIdx.x & 31;
    long row = (long)blockIdx.x * 8 + wave;
    float* p = ms + row * 128;
    float v[4]; float s = 0.f;
#pragma unroll
    for (int k = 0; k < 4; ++k) { v[k] = p[lane + k * 32]; s += v[k]; }
    s = wave_sum(s);
    float mean = s * (1.f / 128.f);
    float q = 0.f;
#pragma unroll
    for (int k = 0; k < 4; ++k) { float d = v[k] - mean; q += d * d; }
    q = wave_sum(q);
    float rstd = rsqrtf(q * (1.f / 128.f) + 1e-5f);
#pragma unroll
    for (int k = 0; k < 4; ++k) {
        int c = lane + k * 32;
        float o = (v[k] - mean) * rstd * g[c] + bb[c];
        p[c] = fmaxf(o, 0.f);
    }
}

// ---------------------------------------------------------------------------
// scores = tanh_feats @ p2_W + p2_b  (Nc=1 GEMV)
// ---------------------------------------------------------------------------
__global__ void score2(const float* __restrict__ tb, const float* __restrict__ w,
                       const float* __restrict__ b2, float* __restrict__ sc) {
    int r = blockIdx.x * BDIM + threadIdx.x;   // 8192
    const float* tr = tb + (long)r * 64;
    float s = b2[0];
#pragma unroll
    for (int k = 0; k < 64; ++k) s += tr[k] * w[k];
    sc[r] = s;
}

// ---------------------------------------------------------------------------
// Attention pooling per batch: softmax over N, pooled[c] = sum w_n * ms[n,c]
// ---------------------------------------------------------------------------
__global__ __launch_bounds__(BDIM) void pool_kernel(const float* __restrict__ ms,
                                                    const float* __restrict__ sc,
                                                    float* __restrict__ pooled) {
    __shared__ float red[BDIM];
    int b = blockIdx.x, tid = threadIdx.x;
    const float* s = sc + b * 1024;
    float lm = -__builtin_inff();
    for (int n = tid; n < 1024; n += BDIM) lm = fmaxf(lm, s[n]);
    red[tid] = lm; __syncthreads();
    for (int st = 128; st > 0; st >>= 1) { if (tid < st) red[tid] = fmaxf(red[tid], red[tid + st]); __syncthreads(); }
    float mx = red[0]; __syncthreads();
    float ls = 0.f;
    for (int n = tid; n < 1024; n += BDIM) ls += __expf(s[n] - mx);
    red[tid] = ls; __syncthreads();
    for (int st = 128; st > 0; st >>= 1) { if (tid < st) red[tid] += red[tid + st]; __syncthreads(); }
    float Z = red[0]; __syncthreads();
    int c = tid & 127, half = tid >> 7;
    float acc = 0.f;
    for (int n = half * 512; n < half * 512 + 512; ++n)
        acc += (__expf(s[n] - mx) / Z) * ms[((long)b * 1024 + n) * 128 + c];
    red[tid] = acc; __syncthreads();
    if (tid < 128) pooled[b * 128 + tid] = red[tid] + red[tid + 128];
}

// ---------------------------------------------------------------------------
// emb = LayerNorm(pooled @ proj_W + proj_b); one block per batch, 256 cols
// ---------------------------------------------------------------------------
__global__ __launch_bounds__(BDIM) void final_kernel(const float* __restrict__ pooled,
                                                     const float* __restrict__ W,
                                                     const float* __restrict__ bb,
                                                     const float* __restrict__ g,
                                                     const float* __restrict__ be,
                                                     float* __restrict__ out) {
    __shared__ float red[BDIM];
    int b = blockIdx.x, c = threadIdx.x;
    const float* p = pooled + b * 128;
    float v = bb[c];
#pragma unroll
    for (int k = 0; k < 128; ++k) v += p[k] * W[k * 256 + c];
    red[c] = v; __syncthreads();
    for (int st = 128; st > 0; st >>= 1) { if (c < st) red[c] += red[c + st]; __syncthreads(); }
    float mean = red[0] * (1.f / 256.f); __syncthreads();
    float d = v - mean;
    red[c] = d * d; __syncthreads();
    for (int st = 128; st > 0; st >>= 1) { if (c < st) red[c] += red[c + st]; __syncthreads(); }
    float rstd = rsqrtf(red[0] * (1.f / 256.f) + 1e-5f);
    out[b * 256 + c] = (v - mean) * rstd * g[c] + be[c];
}

// ---------------------------------------------------------------------------
// Host orchestration
// ---------------------------------------------------------------------------
extern "C" void kernel_launch(void* const* d_in, const int* in_sizes, int n_in,
                              void* d_out, int out_size, void* d_ws, size_t ws_size,
                              hipStream_t stream) {
    (void)in_sizes; (void)n_in; (void)out_size; (void)ws_size;
    const int B = 8, N = 1024, HID = 128, H = 4;

    const float* fc      = (const float*)d_in[0];
    const float* node_W  = (const float*)d_in[1];
    const float* node_b  = (const float*)d_in[2];
    const float* gat_W   = (const float*)d_in[3];
    const float* gat_a   = (const float*)d_in[4];
    const float* ls_W    = (const float*)d_in[5];
    const float* ls_b    = (const float*)d_in[6];
    const float* gs_W    = (const float*)d_in[7];
    const float* gs_b    = (const float*)d_in[8];
    const float* ms_g    = (const float*)d_in[9];
    const float* ms_bv   = (const float*)d_in[10];
    const float* p1_W    = (const float*)d_in[11];
    const float* p1_b    = (const float*)d_in[12];
    const float* p2_W    = (const float*)d_in[13];
    const float* p2_b    = (const float*)d_in[14];
    const float* proj_W  = (const float*)d_in[15];
    const float* proj_b  = (const float*)d_in[16];
    const float* ln_g    = (const float*)d_in[17];
    const float* ln_b    = (const float*)d_in[18];
    float* out = (float*)d_out;

    char* ws = (char*)d_ws;
    size_t off = 0;
    auto alloc = [&](size_t bytes) -> float* {
        float* p = (float*)(ws + off);
        off += (bytes + 255) & ~(size_t)255;
        return p;
    };
    const size_t BNH = (size_t)B * N * HID * 4;       // 4 MB
    float* x      = alloc(BNH);
    float* hbuf   = alloc(BNH);
    float* y1     = alloc(BNH);
    float* y2     = alloc(BNH);
    float* msb    = alloc(BNH);
    float* tbuf   = alloc((size_t)B * N * 64 * 4);
    float* esrc   = alloc((size_t)B * H * N * 4);
    float* edst   = alloc((size_t)B * H * N * 4);
    float* mrow   = alloc((size_t)B * H * N * 4);
    float* invZ   = alloc((size_t)B * H * N * 4);
    float* scores = alloc((size_t)B * N * 4);
    float* invrs  = alloc((size_t)B * N * 4);
    float* pooled = alloc((size_t)B * HID * 4);
    float* Wcat   = alloc((size_t)HID * HID * 4);

    auto gemm = [&](const float* A, const float* Bm, const float* bias, float* C,
                    int M, int K, int Nc, int lda, int ldb, int ldc,
                    long sA, long sB, long sC, const float* rsc, long sRS,
                    int act, int batch) {
        dim3 g(M / 128, Nc / 64, batch);
        wmma_gemm<<<g, BDIM, 0, stream>>>(A, Bm, bias, C, K, lda, ldb, ldc,
                                          sA, sB, sC, rsc, sRS, act);
    };

    // x = relu(fc @ node_W + node_b)   [flatten B into M: fc rows contiguous]
    gemm(fc, node_W, node_b, x, B * N, N, HID, N, HID, HID, 0, 0, 0, nullptr, 0, 1, 1);
    row_inv<<<1024, BDIM, 0, stream>>>(fc, invrs);

    for (int l = 0; l < 3; ++l) {
        pack_wcat<<<64, BDIM, 0, stream>>>(gat_W + (size_t)l * H * HID * 32, Wcat);
        // h = x @ Wcat  -> hbuf [B,N, H*32]
        gemm(x, Wcat, nullptr, hbuf, B * N, HID, HID, HID, HID, HID, 0, 0, 0, nullptr, 0, 0, 1);
        gat_scores<<<128, BDIM, 0, stream>>>(hbuf, gat_a + (size_t)l * H * 64, esrc, edst);
        gat_stats<<<4096, BDIM, 0, stream>>>(fc, esrc, edst, mrow, invZ);
        dim3 ga(N / 128, H, B);
        gat_attend<<<ga, BDIM, 0, stream>>>(fc, hbuf, esrc, edst, mrow, invZ, x);
    }

    // y1 = adj_norm @ x = rowscale(fc @ x); y2 = adj_norm @ y1 (= adj2 @ x)
    gemm(fc, x,  nullptr, y1, N, N, HID, N, HID, HID,
         (long)N * N, (long)N * HID, (long)N * HID, invrs, N, 0, B);
    gemm(fc, y1, nullptr, y2, N, N, HID, N, HID, HID,
         (long)N * N, (long)N * HID, (long)N * HID, invrs, N, 0, B);

    // ms = [y1 @ ls_W + ls_b | y2 @ gs_W + gs_b]
    gemm(y1, ls_W, ls_b, msb,      B * N, HID, 64, HID, 64, HID, 0, 0, 0, nullptr, 0, 0, 1);
    gemm(y2, gs_W, gs_b, msb + 64, B * N, HID, 64, HID, 64, HID, 0, 0, 0, nullptr, 0, 0, 1);
    ln_relu<<<1024, BDIM, 0, stream>>>(msb, ms_g, ms_bv);

    // attention pooling
    gemm(msb, p1_W, p1_b, tbuf, B * N, HID, 64, HID, 64, 64, 0, 0, 0, nullptr, 0, 2, 1);
    score2<<<32, BDIM, 0, stream>>>(tbuf, p2_W, p2_b, scores);
    pool_kernel<<<B, BDIM, 0, stream>>>(msb, scores, pooled);

    // projection + layernorm
    final_kernel<<<B, BDIM, 0, stream>>>(pooled, proj_W, proj_b, ln_g, ln_b, out);
}
